// OuterProductMean_40501541601317
// MI455X (gfx1250) — matrix operand order repaired and checked by reading, not compiled, not run on hardware
//
#include <hip/hip_runtime.h>

typedef __attribute__((ext_vector_type(16))) __bf16 v16bf;
typedef __attribute__((ext_vector_type(8)))  float  v8f;

union Frag { v16bf v; uint4 q[2]; };

__device__ __forceinline__ unsigned short f2bf(float f) {
    unsigned int u = __float_as_uint(f);
    unsigned int r = u + 0x7FFFu + ((u >> 16) & 1u);   // round-to-nearest-even
    return (unsigned short)(r >> 16);
}

__device__ __forceinline__ unsigned int pack2bf(float lo, float hi) {
    return (unsigned int)f2bf(lo) | ((unsigned int)f2bf(hi) << 16);
}

__device__ __forceinline__ float wave_sum(float v) {
#pragma unroll
    for (int o = 16; o > 0; o >>= 1) v += __shfl_xor(v, o, 32);
    return v;
}

// ---------------------------------------------------------------------------
// Wo [1024,128] f32 -> WoT [128][1024] bf16 with PERMUTED K order q' = d*32+c
// (matches the Os LDS layout in k_opm; WMMA reduction order is free).
// ---------------------------------------------------------------------------
__global__ __launch_bounds__(256) void k_wot(const float* __restrict__ Wo,
                                             unsigned short* __restrict__ WoT) {
    int idx = blockIdx.x * 256 + threadIdx.x;   // 0 .. 131071
    int e  = idx >> 10;
    int qp = idx & 1023;        // q' = d*32 + c
    int d  = qp >> 5;
    int c  = qp & 31;
    WoT[idx] = f2bf(Wo[(c * 32 + d) * 128 + e]);
}

// ---------------------------------------------------------------------------
// LayerNorm + a/b projections + mask, write aT/bT as [L][32][N] bf16
// ---------------------------------------------------------------------------
__global__ __launch_bounds__(256) void k_proj(const float* __restrict__ m,
                                              const float* __restrict__ mask,
                                              const float* __restrict__ gamma,
                                              const float* __restrict__ beta,
                                              const float* __restrict__ Wa,
                                              const float* __restrict__ ba,
                                              const float* __restrict__ Wb,
                                              const float* __restrict__ bb,
                                              unsigned short* __restrict__ aT,
                                              unsigned short* __restrict__ bT) {
    __shared__ float WS[8192];          // 256 x 32 weight tile (32 KB)
    __shared__ float xs[8][260];        // per-wave normalized row

    const int tid  = threadIdx.x;
    const int w    = tid >> 5;
    const int lane = tid & 31;
    const int row  = blockIdx.x * 8 + w;      // 0..32767
    const int n    = row >> 8;
    const int l    = row & 255;

    for (int idx = tid; idx < 8192; idx += 256) WS[idx] = Wa[idx];

    const float* mrow = m + (size_t)row * 256;
    float4 v0 = ((const float4*)mrow)[lane * 2 + 0];
    float4 v1 = ((const float4*)mrow)[lane * 2 + 1];
    float s  = v0.x + v0.y + v0.z + v0.w + v1.x + v1.y + v1.z + v1.w;
    float s2 = v0.x*v0.x + v0.y*v0.y + v0.z*v0.z + v0.w*v0.w
             + v1.x*v1.x + v1.y*v1.y + v1.z*v1.z + v1.w*v1.w;
    s  = wave_sum(s);
    s2 = wave_sum(s2);
    const float mean = s * (1.0f / 256.0f);
    const float var  = s2 * (1.0f / 256.0f) - mean * mean;
    const float rstd = rsqrtf(var + 1e-5f);

    float4 g0 = ((const float4*)gamma)[lane * 2 + 0];
    float4 g1 = ((const float4*)gamma)[lane * 2 + 1];
    float4 b0 = ((const float4*)beta )[lane * 2 + 0];
    float4 b1 = ((const float4*)beta )[lane * 2 + 1];
    const int cb = lane * 8;
    xs[w][cb + 0] = (v0.x - mean) * rstd * g0.x + b0.x;
    xs[w][cb + 1] = (v0.y - mean) * rstd * g0.y + b0.y;
    xs[w][cb + 2] = (v0.z - mean) * rstd * g0.z + b0.z;
    xs[w][cb + 3] = (v0.w - mean) * rstd * g0.w + b0.w;
    xs[w][cb + 4] = (v1.x - mean) * rstd * g1.x + b1.x;
    xs[w][cb + 5] = (v1.y - mean) * rstd * g1.y + b1.y;
    xs[w][cb + 6] = (v1.z - mean) * rstd * g1.z + b1.z;
    xs[w][cb + 7] = (v1.w - mean) * rstd * g1.w + b1.w;
    __syncthreads();

    float aacc = ba[lane];
    for (int c = 0; c < 256; ++c)
        aacc = fmaf(xs[w][c], WS[c * 32 + lane], aacc);
    __syncthreads();

    for (int idx = tid; idx < 8192; idx += 256) WS[idx] = Wb[idx];
    __syncthreads();

    float bacc = bb[lane];
    for (int c = 0; c < 256; ++c)
        bacc = fmaf(xs[w][c], WS[c * 32 + lane], bacc);

    const float mk = mask[n * 256 + l];
    aT[(size_t)(l * 32 + lane) * 128 + n] = f2bf(aacc * mk);
    bT[(size_t)(l * 32 + lane) * 128 + n] = f2bf(bacc * mk);
}

// ---------------------------------------------------------------------------
// Fused outer-product-mean + Wo projection (2 i x 16 j = 32 pairs / block).
// Stage A: O[2i][32c][512jd] over K=N=128 (bf16 WMMA, aT staged in LDS).
// Scale, pack to bf16, park in LDS as Os[p][d][c] (d-major K permutation,
// packed b128 writes). Stage B: z[32 x 128] = Os @ WoT' with each WoT
// fragment reused across both M-tiles.
// ---------------------------------------------------------------------------
#define AS_PITCH  136   // 128+8 shorts: 68-dword row stride -> 4-bank lane shift
#define OS_DPITCH 40    // 32+8 shorts per d-row: 20-dword stride -> 64 banks once
#define OS_PPITCH 1288  // 32*40+8 shorts per pair: 644-dword stride -> 4-bank shift

__global__ __launch_bounds__(256) void k_opm(const unsigned short* __restrict__ aT,
                                             const unsigned short* __restrict__ bT,
                                             const unsigned short* __restrict__ WoT,
                                             const float* __restrict__ mask,
                                             const float* __restrict__ bo,
                                             float* __restrict__ out) {
    __shared__ unsigned short aS[64 * AS_PITCH];    // ~17 KB
    __shared__ unsigned short Os[32 * OS_PPITCH];   // ~80.5 KB

    const int i0   = blockIdx.x * 2;    // 0..254
    const int j0   = blockIdx.y * 16;   // 0..240
    const int tid  = threadIdx.x;
    const int w    = tid >> 5;
    const int lane = tid & 31;
    const int lh   = lane & 15;
    const bool hi  = lane >= 16;

    // --- stage both aT rows into LDS (coalesced b128 copy, padded pitch) ---
    {
        const uint4* src = (const uint4*)(aT + (size_t)(i0 * 32) * 128); // 1024 x uint4
#pragma unroll
        for (int k2 = 0; k2 < 4; ++k2) {
            int idx = tid + k2 * 256;          // 0..1023
            int row = idx >> 4, seg = idx & 15;
            *(uint4*)&aS[row * AS_PITCH + seg * 8] = src[idx];
        }
    }

    // --- pair-mask scales: 4 per wave (2 i's x 2 j's) ----------------------
    const int jA = j0 + 2 * w, jB = jA + 1;
    float pm[2][2] = {{0.f, 0.f}, {0.f, 0.f}};
#pragma unroll
    for (int t = 0; t < 4; ++t) {
        int n = lane + t * 32;
        float mi0 = mask[n * 256 + i0];
        float mi1 = mask[n * 256 + i0 + 1];
        float mj0 = mask[n * 256 + jA];
        float mj1 = mask[n * 256 + jB];
        pm[0][0] += mi0 * mj0;  pm[0][1] += mi0 * mj1;
        pm[1][0] += mi1 * mj0;  pm[1][1] += mi1 * mj1;
    }
    float sc[2][2];
#pragma unroll
    for (int it = 0; it < 2; ++it)
#pragma unroll
        for (int jj = 0; jj < 2; ++jj)
            sc[it][jj] = __builtin_amdgcn_rcpf(wave_sum(pm[it][jj]) + 1e-8f);

    __syncthreads();   // aS visible to all waves

    // --- base pointers (all K-loop accesses become immediate offsets) ------
    const unsigned short* paA = &aS[lh * AS_PITCH + (hi ? 8 : 0)];
    const unsigned short* pbB[4];
#pragma unroll
    for (int nt = 0; nt < 4; ++nt) {
        const int jd = w * 64 + nt * 16 + lh;
        const int j  = j0 + (jd >> 5);
        const int d  = jd & 31;
        pbB[nt] = bT + (size_t)(j * 32 + d) * 128 + (hi ? 8 : 0);
    }

    // --- Stage A -----------------------------------------------------------
    v8f acc[2][2][4];          // [it][mt(c-tile)][nt(jd-tile)]
#pragma unroll
    for (int it = 0; it < 2; ++it)
#pragma unroll
        for (int mt = 0; mt < 2; ++mt)
#pragma unroll
            for (int nt = 0; nt < 4; ++nt)
#pragma unroll
                for (int r = 0; r < 8; ++r) acc[it][mt][nt][r] = 0.0f;

#pragma unroll
    for (int kk = 0; kk < 4; ++kk) {            // 32-wide chunks of K=N=128
        Frag A[2][2];
#pragma unroll
        for (int it = 0; it < 2; ++it)
#pragma unroll
            for (int mt = 0; mt < 2; ++mt) {
                const unsigned short* pa = paA + (it * 32 + mt * 16) * AS_PITCH + kk * 32;
                A[it][mt].q[0] = *(const uint4*)(pa);
                A[it][mt].q[1] = *(const uint4*)(pa + 16);
            }
        Frag Bf[4];
#pragma unroll
        for (int nt = 0; nt < 4; ++nt) {        // grouped -> one load clause
            const unsigned short* pb = pbB[nt] + kk * 32;
            Bf[nt].q[0] = *(const uint4*)(pb);
            Bf[nt].q[1] = *(const uint4*)(pb + 16);
        }
#pragma unroll
        for (int nt = 0; nt < 4; ++nt)
#pragma unroll
            for (int it = 0; it < 2; ++it)
#pragma unroll
                for (int mt = 0; mt < 2; ++mt)
                    acc[it][mt][nt] = __builtin_amdgcn_wmma_f32_16x16x32_bf16(
                        false, A[it][mt].v, false, Bf[nt].v, (short)0, acc[it][mt][nt], false, false);
    }

    // --- scale + park O in LDS: Os[p][d][c], packed b128 writes ------------
    {
        unsigned short* pw0 = &Os[(2 * w) * OS_PPITCH + lh * OS_DPITCH + (hi ? 8 : 0)];
#pragma unroll
        for (int it = 0; it < 2; ++it)
#pragma unroll
            for (int nt = 0; nt < 4; ++nt) {
                const float s = sc[it][nt >> 1];
#pragma unroll
                for (int mt = 0; mt < 2; ++mt) {
                    const v8f a = acc[it][mt][nt];
                    uint4 pk;
                    pk.x = pack2bf(a[0] * s, a[1] * s);
                    pk.y = pack2bf(a[2] * s, a[3] * s);
                    pk.z = pack2bf(a[4] * s, a[5] * s);
                    pk.w = pack2bf(a[6] * s, a[7] * s);
                    // p = it*16 + 2w + (nt>>1); d = (nt&1)*16 + lh; c0 = mt*16 + hi*8
                    *(uint4*)(pw0 + (it * 16 + (nt >> 1)) * OS_PPITCH
                                  + (nt & 1) * 16 * OS_DPITCH + mt * 16) = pk;
                }
            }
    }
    __syncthreads();

    // --- Stage B: z[p=0..31][e = w*16 + lh], K = q' = 1024 (d-major) -------
    v8f z2[2];
#pragma unroll
    for (int it = 0; it < 2; ++it)
#pragma unroll
        for (int r = 0; r < 8; ++r) z2[it][r] = 0.0f;

    const int e = w * 16 + lh;
    const unsigned short* pwB = WoT + (size_t)e * 1024 + (hi ? 8 : 0);
    const unsigned short* plA = &Os[lh * OS_PPITCH + (hi ? 8 : 0)];

#pragma unroll
    for (int d = 0; d < 32; ++d) {              // K chunk d covers q' = d*32 .. d*32+31
        Frag Bf;
        Bf.q[0] = *(const uint4*)(pwB + d * 32);
        Bf.q[1] = *(const uint4*)(pwB + d * 32 + 16);
#pragma unroll
        for (int it = 0; it < 2; ++it) {        // reuse WoT frag for both M-tiles
            Frag Af;
            const unsigned short* pl = plA + it * 16 * OS_PPITCH + d * OS_DPITCH;
            Af.q[0] = *(const uint4*)(pl);
            Af.q[1] = *(const uint4*)(pl + 16);
            z2[it] = __builtin_amdgcn_wmma_f32_16x16x32_bf16(
                false, Af.v, false, Bf.v, (short)0, z2[it], false, false);
        }
    }

    const float bv = bo[e];
    float* pout = out + (size_t)((i0 * 256) + j0 + (hi ? 8 : 0)) * 128 + e;
#pragma unroll
    for (int it = 0; it < 2; ++it)
#pragma unroll
        for (int r = 0; r < 8; ++r)             // row within M-tile = j_local
            pout[it * (256 * 128) + r * 128] = z2[it][r] + bv;
}

// ---------------------------------------------------------------------------
extern "C" void kernel_launch(void* const* d_in, const int* in_sizes, int n_in,
                              void* d_out, int out_size, void* d_ws, size_t ws_size,
                              hipStream_t stream) {
    const float* m    = (const float*)d_in[0];
    const float* mask = (const float*)d_in[1];
    const float* g    = (const float*)d_in[2];
    const float* beta = (const float*)d_in[3];
    const float* Wa   = (const float*)d_in[4];
    const float* ba   = (const float*)d_in[5];
    const float* Wb   = (const float*)d_in[6];
    const float* bb   = (const float*)d_in[7];
    const float* Wo   = (const float*)d_in[8];
    const float* bo   = (const float*)d_in[9];
    float* out = (float*)d_out;

    // workspace: aT/bT [L=256][32][N=128] bf16 (2 MB each), WoT [128][1024] bf16
    unsigned short* aT  = (unsigned short*)d_ws;
    unsigned short* bT  = aT + (size_t)256 * 32 * 128;
    unsigned short* WoT = bT + (size_t)256 * 32 * 128;

    k_wot <<<512, 256, 0, stream>>>(Wo, WoT);
    k_proj<<<4096, 256, 0, stream>>>(m, mask, g, beta, Wa, ba, Wb, bb, aT, bT);
    k_opm <<<dim3(128, 16), 256, 0, stream>>>(aT, bT, WoT, mask, bo, out);
}
